// ntm_plus_memory_5488968204594
// MI455X (gfx1250) — matrix operand order repaired and checked by reading, not compiled, not run on hardware
//
#include <hip/hip_runtime.h>
#include <hip/hip_bf16.h>
#include <math.h>

// Problem constants
#define B_      16384
#define C_      256
#define N_      128
#define M_      20
#define EMIT_   26
#define PSTRIDE 32        // padded per-row param stride in workspace

typedef __attribute__((ext_vector_type(16))) __bf16 v16bf;
typedef __attribute__((ext_vector_type(16))) float  v16f;
typedef __attribute__((ext_vector_type(8)))  float  v8f;

union BFrag  { v16bf v; unsigned short u[16]; };
union U16x8  { uint4 q; unsigned short u[8];  };

__device__ __forceinline__ unsigned short f2bf(float f) {
  unsigned int x = __float_as_uint(f);
  x += 0x7FFFu + ((x >> 16) & 1u);     // round-to-nearest-even
  return (unsigned short)(x >> 16);
}
__device__ __forceinline__ float softplusf(float x) {
  return fmaxf(x, 0.f) + log1pf(expf(-fabsf(x)));
}
__device__ __forceinline__ float sigmoidf(float x) {
  return 1.f / (1.f + expf(-x));
}

// ---------------------------------------------------------------------------
// Kernel A: fc = tanh(co @ W_addr + b) via bf16 WMMA, then per-row params.
// Block = 128 threads (4 waves). Each wave: 16 rows x 32 cols (EMIT padded).
// Grid = B/64.
// ---------------------------------------------------------------------------
__global__ __launch_bounds__(128)
void ntm_fc_kernel(const float* __restrict__ co,
                   const float* __restrict__ W_addr,
                   const float* __restrict__ b_addr,
                   float* __restrict__ params) {
  __shared__ unsigned short sW[32 * C_];   // W_addr, col-major bf16, padded to 32 cols (16 KB)
  __shared__ float sFC[64 * 32];           // fc results for this block (8 KB)

  const int tid  = threadIdx.x;
  const int wave = tid >> 5;
  const int lane = tid & 31;
  const int half = lane >> 4;   // lane-half selects K sub-range
  const int l16  = lane & 15;

  // Stage W_addr -> LDS, col-major bf16, zero-padded cols [26,32)
  for (int i = tid; i < 32 * C_; i += 128) {
    int col = i >> 8;            // 0..31
    int k   = i & (C_ - 1);      // 0..255
    float v = (col < EMIT_) ? W_addr[k * EMIT_ + col] : 0.f;
    sW[col * C_ + k] = f2bf(v);
  }
  __syncthreads();

  const int    row  = blockIdx.x * 64 + wave * 16 + l16;   // A-matrix row for this lane
  const float* rowp = co + (size_t)row * C_;

  v8f acc0 = {};   // cols 0..15
  v8f acc1 = {};   // cols 16..31

  for (int k0 = 0; k0 < C_; k0 += 32) {
    // ---- A fragment: 16-bit A layout -> lane-half holds K runs
    //      {h*8 .. h*8+7} and {16+h*8 .. 16+h*8+7}, each a contiguous f32 run.
    BFrag a;
    {
      const float4* p0 = (const float4*)(rowp + k0 + half * 8);
      const float4* p1 = (const float4*)(rowp + k0 + 16 + half * 8);
      float4 x0 = p0[0], x1 = p0[1], x2 = p1[0], x3 = p1[1];
      v16f af;
      af[0]  = x0.x; af[1]  = x0.y; af[2]  = x0.z; af[3]  = x0.w;
      af[4]  = x1.x; af[5]  = x1.y; af[6]  = x1.z; af[7]  = x1.w;
      af[8]  = x2.x; af[9]  = x2.y; af[10] = x2.z; af[11] = x2.w;
      af[12] = x3.x; af[13] = x3.y; af[14] = x3.z; af[15] = x3.w;
      a.v = __builtin_convertvector(af, v16bf);   // -> v_cvt_pk_bf16_f32 pairs
    }
    // ---- B fragments: lane-half holds K runs {h*8..h*8+7} and {16+h*8..16+h*8+7} for column l16
    BFrag b0, b1;
    {
      const unsigned short* wp0 = &sW[(l16)      * C_ + k0 + half * 8];
      const unsigned short* wp1 = &sW[(16 + l16) * C_ + k0 + half * 8];
      U16x8 t;
      t.q = *(const uint4*)(wp0);
      #pragma unroll
      for (int j = 0; j < 8; ++j) b0.u[j] = t.u[j];
      t.q = *(const uint4*)(wp0 + 16);
      #pragma unroll
      for (int j = 0; j < 8; ++j) b0.u[8 + j] = t.u[j];
      t.q = *(const uint4*)(wp1);
      #pragma unroll
      for (int j = 0; j < 8; ++j) b1.u[j] = t.u[j];
      t.q = *(const uint4*)(wp1 + 16);
      #pragma unroll
      for (int j = 0; j < 8; ++j) b1.u[8 + j] = t.u[j];
    }
    acc0 = __builtin_amdgcn_wmma_f32_16x16x32_bf16(false, a.v, false, b0.v,
                                                   (short)0, acc0, false, false);
    acc1 = __builtin_amdgcn_wmma_f32_16x16x32_bf16(false, a.v, false, b1.v,
                                                   (short)0, acc1, false, false);
  }

  // ---- Scatter C/D (32-bit 16x16 layout: VGPR r -> M = r + 8*half, N = l16)
  const int mbase = wave * 16 + half * 8;
  #pragma unroll
  for (int r = 0; r < 8; ++r) {
    int rr = mbase + r;
    sFC[rr * 32 + l16] = tanhf(acc0[r] + b_addr[l16]);
    float bb = (16 + l16 < EMIT_) ? b_addr[16 + l16] : 0.f;
    sFC[rr * 32 + 16 + l16] = tanhf(acc1[r] + bb);
  }
  __syncthreads();

  // ---- Per-row addressing parameters (one thread per row)
  if (tid < 64) {
    const float* fc = &sFC[tid * 32];
    float* P = params + ((size_t)blockIdx.x * 64 + tid) * PSTRIDE;
    float ksq = 0.f;
    #pragma unroll
    for (int m = 0; m < M_; ++m) { float kv = fc[m]; ksq = fmaf(kv, kv, ksq); P[m] = kv; }
    float kmag = sqrtf(fmaxf(ksq, 1e-30f));
    float beta = softplusf(fc[20]);
    P[20] = beta / kmag;                     // fold beta/|k| into one scale
    P[21] = sigmoidf(fc[21]);                // g
    float s0 = fc[22], s1 = fc[23], s2 = fc[24];
    float sm = fmaxf(s0, fmaxf(s1, s2));
    float e0 = expf(s0 - sm), e1 = expf(s1 - sm), e2 = expf(s2 - sm);
    float inv = 1.f / (e0 + e1 + e2);
    P[22] = e0 * inv; P[23] = e1 * inv; P[24] = e2 * inv;   // shift softmax
    P[25] = 1.f + softplusf(fc[25]);         // gamma
  }
}

// ---------------------------------------------------------------------------
// Kernel B: content addressing + interpolate + circular shift + sharpen.
// One block (128 threads) per batch element; thread = memory slot n.
// Streams M_t[b] (10 KB) once with float4 loads. Grid = B.
// ---------------------------------------------------------------------------
__global__ __launch_bounds__(128)
void ntm_addr_kernel(const float* __restrict__ Mt,
                     const float* __restrict__ w_prev,
                     const float* __restrict__ params,
                     float* __restrict__ out) {
  __shared__ float sp[PSTRIDE];
  __shared__ float red[N_];
  __shared__ float swg[N_];

  const int b   = blockIdx.x;
  const int tid = threadIdx.x;

  if (tid < PSTRIDE) sp[tid] = params[(size_t)b * PSTRIDE + tid];
  __syncthreads();

  // dot(k, M[b,n,:]) and |M[b,n,:]|^2
  const float* Mrow = Mt + (size_t)b * (N_ * M_) + tid * M_;
  float dot = 0.f, msq = 0.f;
  #pragma unroll
  for (int m = 0; m < M_; m += 4) {
    float4 v = *(const float4*)(Mrow + m);
    dot = fmaf(v.x, sp[m],     dot); dot = fmaf(v.y, sp[m + 1], dot);
    dot = fmaf(v.z, sp[m + 2], dot); dot = fmaf(v.w, sp[m + 3], dot);
    msq = fmaf(v.x, v.x, msq);       msq = fmaf(v.y, v.y, msq);
    msq = fmaf(v.z, v.z, msq);       msq = fmaf(v.w, v.w, msq);
  }
  float score = sp[20] * dot * rsqrtf(fmaxf(msq, 1e-30f));

  // softmax over n (block reduce: max, then sum)
  red[tid] = score; __syncthreads();
  for (int off = 64; off > 0; off >>= 1) {
    if (tid < off) red[tid] = fmaxf(red[tid], red[tid + off]);
    __syncthreads();
  }
  float mx = red[0]; __syncthreads();
  float ex = expf(score - mx);
  red[tid] = ex; __syncthreads();
  for (int off = 64; off > 0; off >>= 1) {
    if (tid < off) red[tid] += red[tid + off];
    __syncthreads();
  }
  float wc = ex / red[0];
  __syncthreads();

  // interpolate with previous weights
  float g  = sp[21];
  float wp = w_prev[(size_t)b * N_ + tid];
  float wg = fmaf(g, wc - wp, wp);
  swg[tid] = wg;
  __syncthreads();

  // circular shift (S=3): w_tilde[n] = wg[n+1]*s0 + wg[n]*s1 + wg[n-1]*s2
  float wt = swg[(tid + 1) & (N_ - 1)] * sp[22]
           + wg                        * sp[23]
           + swg[(tid + N_ - 1) & (N_ - 1)] * sp[24];

  // sharpen + normalize
  float wr = powf(fmaxf(wt, 1e-30f), sp[25]);
  red[tid] = wr; __syncthreads();
  for (int off = 64; off > 0; off >>= 1) {
    if (tid < off) red[tid] += red[tid + off];
    __syncthreads();
  }
  out[(size_t)b * N_ + tid] = wr / red[0];
}

// ---------------------------------------------------------------------------
extern "C" void kernel_launch(void* const* d_in, const int* in_sizes, int n_in,
                              void* d_out, int out_size, void* d_ws, size_t ws_size,
                              hipStream_t stream) {
  (void)in_sizes; (void)n_in; (void)out_size; (void)ws_size;
  const float* co     = (const float*)d_in[0];   // (B, C)
  const float* wprev  = (const float*)d_in[1];   // (B, N)
  const float* Mt     = (const float*)d_in[2];   // (B, N, M)
  const float* W_addr = (const float*)d_in[3];   // (C, EMIT)
  const float* b_addr = (const float*)d_in[4];   // (EMIT,)
  float*       out    = (float*)d_out;           // (B, N)
  float*       params = (float*)d_ws;            // B * PSTRIDE floats (2 MB)

  ntm_fc_kernel<<<B_ / 64, 128, 0, stream>>>(co, W_addr, b_addr, params);
  ntm_addr_kernel<<<B_, 128, 0, stream>>>(Mt, wprev, params, out);
}
